// BoundaryFluxAttention_67705864454353
// MI455X (gfx1250) — compile-verified, matrix operand order checked by reading, not run on hardware
//
#include <hip/hip_runtime.h>
#include <hip/hip_bf16.h>
#include <cstdint>
#include <cstddef>

typedef unsigned short u16;
typedef __attribute__((ext_vector_type(16))) __bf16 v16bf;
typedef __attribute__((ext_vector_type(8)))  float  v8f;

#define D_MODEL   1024
#define NUM_HEADS 16
#define HEAD_DIM  64
#define SEQ_T     2048
#define BATCH     2
#define BIAS_COEF 0.1f

union Frag16 { v16bf v; u16 u[16]; };

__device__ __forceinline__ u16 f2bf(float f) {
  unsigned int x = __float_as_uint(f);
  unsigned int r = 0x7FFFu + ((x >> 16) & 1u);   // round-to-nearest-even
  return (u16)((x + r) >> 16);
}

// 16-bit A/B fragment, K-major contiguous source:
// lane half h, element e -> k = (e<8 ? 0 : 16) + 8*h + (e%8)
// rowptr must be 16B aligned; lowers to two ds_load_b128 / global_load_b128.
__device__ __forceinline__ v16bf frag_rowmajor(const u16* rowptr, int h) {
  Frag16 f;
#pragma unroll
  for (int e = 0; e < 8; ++e) f.u[e]     = rowptr[h * 8 + e];
#pragma unroll
  for (int e = 0; e < 8; ++e) f.u[8 + e] = rowptr[16 + h * 8 + e];
  return f.v;
}

#define WMMA_BF16(a, b, c) \
  __builtin_amdgcn_wmma_f32_16x16x32_bf16(false, (a), false, (b), (short)0, (c), false, false)

// ---------------------------------------------------------------- convert (row-major)
__global__ void cvt_f32_bf16(const float* __restrict__ src, u16* __restrict__ dst, int n) {
  int i = blockIdx.x * blockDim.x + threadIdx.x;
  if (i < n) dst[i] = f2bf(src[i]);
}

// ------------------------------------------------- convert + transpose: [K][N] f32 -> [N][K] bf16
__global__ __launch_bounds__(256)
void cvt_transpose(const float* __restrict__ src, u16* __restrict__ dst, int K, int N) {
  __shared__ float tile[32][33];
  int kBase = blockIdx.y * 32, nBase = blockIdx.x * 32;
  int tid = threadIdx.x;
  int r = tid >> 3, c4 = (tid & 7) * 4;
  float4 v = *(const float4*)(src + (size_t)(kBase + r) * N + nBase + c4);
  tile[r][c4 + 0] = v.x; tile[r][c4 + 1] = v.y;
  tile[r][c4 + 2] = v.z; tile[r][c4 + 3] = v.w;
  __syncthreads();
  int n = tid >> 3, k4 = (tid & 7) * 4;
  unsigned long long pk = 0;
#pragma unroll
  for (int j = 0; j < 4; ++j)
    pk |= (unsigned long long)f2bf(tile[k4 + j][n]) << (16 * j);
  *(unsigned long long*)(dst + (size_t)(nBase + n) * K + kBase + k4) = pk;
}

// ---------------------------------------------------------------- QKV GEMM (double-buffered)
// C[4096,3072] = X[4096,1024] @ Wqkv + b ; BT is Wqkv^T [3072][1024] bf16.
// Scatter: Q,K -> [B,H,T,64]; V -> transposed [B,H,64,T].
__global__ __launch_bounds__(256)
void gemm_qkv(const u16* __restrict__ A, const u16* __restrict__ BT,
              const float* __restrict__ bias,
              u16* __restrict__ Qb, u16* __restrict__ Kb, u16* __restrict__ Vtb) {
  const int K = 1024;
  __shared__ alignas(16) u16 As[2][64][40];
  __shared__ alignas(16) u16 Bt[2][128][40];

  int tid = threadIdx.x, lane = tid & 31, w = tid >> 5;
  int hl = lane >> 4, n16 = lane & 15;
  int wm = w & 3, wn = w >> 2;
  int rowBase = blockIdx.y * 64, colBase = blockIdx.x * 128;

  int ar = tid >> 2, ac = (tid & 3) * 8;   // A: 64 rows x 32 k
  int bn = tid >> 1, bk = (tid & 1) * 16;  // B: 128 cols x 32 k

  const u16* aPtr = A  + (size_t)(rowBase + ar) * K + ac;
  const u16* bPtr = BT + (size_t)(colBase + bn) * K + bk;

  // prologue: tile 0 -> buffer 0
  uint4 aR  = *(const uint4*)aPtr;
  uint4 bR0 = *(const uint4*)bPtr;
  uint4 bR1 = *(const uint4*)(bPtr + 8);
  *(uint4*)&As[0][ar][ac]     = aR;
  *(uint4*)&Bt[0][bn][bk]     = bR0;
  *(uint4*)&Bt[0][bn][bk + 8] = bR1;
  __syncthreads();

  v8f acc[4] = {};
  int buf = 0;
  for (int k0 = 0; k0 < K; k0 += 32) {
    int kn = (k0 + 32 < K) ? (k0 + 32) : 0;      // clamped prefetch (last iter redundant)
    aR  = *(const uint4*)(aPtr + kn);
    bR0 = *(const uint4*)(bPtr + kn);
    bR1 = *(const uint4*)(bPtr + kn + 8);

    v16bf a = frag_rowmajor(&As[buf][wm * 16 + n16][0], hl);
#pragma unroll
    for (int nf = 0; nf < 4; ++nf) {
      v16bf bfv = frag_rowmajor(&Bt[buf][wn * 64 + nf * 16 + n16][0], hl);
      acc[nf] = WMMA_BF16(a, bfv, acc[nf]);
    }

    buf ^= 1;
    *(uint4*)&As[buf][ar][ac]     = aR;
    *(uint4*)&Bt[buf][bn][bk]     = bR0;
    *(uint4*)&Bt[buf][bn][bk + 8] = bR1;
    __syncthreads();
  }

  // branchless scatter epilogue
#pragma unroll
  for (int nf = 0; nf < 4; ++nf) {
    int col = colBase + wn * 64 + nf * 16 + n16;
    float bv = bias[col];
    int which = col >> 10, hh = (col >> 6) & 15, d = col & 63;
    u16* base = (which == 0) ? Qb : ((which == 1) ? Kb : Vtb);
#pragma unroll
    for (int j = 0; j < 8; ++j) {
      int row = rowBase + wm * 16 + j + 8 * hl;
      int bb = row >> 11, t = row & (SEQ_T - 1);
      size_t hoff  = (size_t)(bb * NUM_HEADS + hh);
      size_t idxQK = (hoff * SEQ_T + t) * HEAD_DIM + d;
      size_t idxV  = (hoff * HEAD_DIM + d) * SEQ_T + t;
      size_t idx   = (which == 2) ? idxV : idxQK;
      base[idx] = f2bf(acc[nf][j] + bv);
    }
  }
}

// ---------------------------------------------------------------- flash attention (double-buffered)
// Per block: one (b,h), 128 query rows (16/wave). K tiles [key][d], V tiles transposed [d][key].
__global__ __launch_bounds__(256)
void flash_attn(const u16* __restrict__ Qb, const u16* __restrict__ Kb,
                const u16* __restrict__ Vtb, const float* __restrict__ bscore,
                u16* __restrict__ attnb) {
  __shared__ alignas(16) u16 Ks [2][64][72];   // [key][d]
  __shared__ alignas(16) u16 Vts[2][64][72];   // [d][key]
  __shared__ alignas(16) u16 Ps [8][16][72];

  const int QT = SEQ_T / 128, NT = SEQ_T / 64;
  int bh = blockIdx.x / QT, qt = blockIdx.x % QT;
  int b = bh >> 4, hh = bh & 15;

  int tid = threadIdx.x, lane = tid & 31, w = tid >> 5;
  int hl = lane >> 4, n16 = lane & 15;
  int qrow0 = qt * 128 + w * 16;
  const size_t headOff = (size_t)bh * SEQ_T * HEAD_DIM;

  const u16* qrow = Qb + headOff + (size_t)(qrow0 + n16) * HEAD_DIM;
  v16bf q0 = frag_rowmajor(qrow, hl);
  v16bf q1 = frag_rowmajor(qrow + 32, hl);

  float mo[8], li[8];
  v8f o[4] = {};
#pragma unroll
  for (int j = 0; j < 8; ++j) { mo[j] = -1e30f; li[j] = 0.0f; }
  const float scale = 0.125f;  // 1/sqrt(64)

  int lr = tid >> 2, lc = (tid & 3) * 16;  // stage: 64 rows x 64 cols, 16 elems/thread
  const u16* kPtr = Kb  + headOff + (size_t)lr * HEAD_DIM + lc;
  const u16* vPtr = Vtb + headOff + (size_t)lr * SEQ_T + lc;

  // prologue: tile 0 -> buffer 0
  uint4 kRa = *(const uint4*)kPtr;
  uint4 kRb = *(const uint4*)(kPtr + 8);
  uint4 vRa = *(const uint4*)vPtr;
  uint4 vRb = *(const uint4*)(vPtr + 8);
  *(uint4*)&Ks[0][lr][lc]      = kRa;
  *(uint4*)&Ks[0][lr][lc + 8]  = kRb;
  *(uint4*)&Vts[0][lr][lc]     = vRa;
  *(uint4*)&Vts[0][lr][lc + 8] = vRb;
  __syncthreads();

  int buf = 0;
  for (int kt = 0; kt < NT; ++kt) {
    int ktn = (kt + 1 < NT) ? (kt + 1) : 0;  // clamped prefetch
    kRa = *(const uint4*)(kPtr + (size_t)ktn * 64 * HEAD_DIM);
    kRb = *(const uint4*)(kPtr + (size_t)ktn * 64 * HEAD_DIM + 8);
    vRa = *(const uint4*)(vPtr + ktn * 64);
    vRb = *(const uint4*)(vPtr + ktn * 64 + 8);

    // S = (Q K^T) * scale + boundary bias
    v8f s[4];
#pragma unroll
    for (int nf = 0; nf < 4; ++nf) {
      v8f a = {};
      a = WMMA_BF16(q0, frag_rowmajor(&Ks[buf][nf * 16 + n16][0], hl), a);
      a = WMMA_BF16(q1, frag_rowmajor(&Ks[buf][nf * 16 + n16][32], hl), a);
      float bias = bscore[b * SEQ_T + kt * 64 + nf * 16 + n16] * BIAS_COEF;
#pragma unroll
      for (int j = 0; j < 8; ++j) s[nf][j] = a[j] * scale + bias;
    }

    // online softmax (row = j + 8*hl; columns across the 16-lane half)
#pragma unroll
    for (int j = 0; j < 8; ++j) {
      float m_ = fmaxf(fmaxf(s[0][j], s[1][j]), fmaxf(s[2][j], s[3][j]));
      m_ = fmaxf(m_, __shfl_xor(m_, 1, 32));
      m_ = fmaxf(m_, __shfl_xor(m_, 2, 32));
      m_ = fmaxf(m_, __shfl_xor(m_, 4, 32));
      m_ = fmaxf(m_, __shfl_xor(m_, 8, 32));
      float mn = fmaxf(mo[j], m_);
      float corr = __expf(mo[j] - mn);
      mo[j] = mn;
      li[j] *= corr;
#pragma unroll
      for (int nf = 0; nf < 4; ++nf) o[nf][j] *= corr;
      float p0 = __expf(s[0][j] - mn); s[0][j] = p0;
      float p1 = __expf(s[1][j] - mn); s[1][j] = p1;
      float p2 = __expf(s[2][j] - mn); s[2][j] = p2;
      float p3 = __expf(s[3][j] - mn); s[3][j] = p3;
      float rs = p0 + p1 + p2 + p3;
      rs += __shfl_xor(rs, 1, 32);
      rs += __shfl_xor(rs, 2, 32);
      rs += __shfl_xor(rs, 4, 32);
      rs += __shfl_xor(rs, 8, 32);
      li[j] += rs;
    }

    // restage P (D-layout) -> LDS (row-major in key) -> A fragments
#pragma unroll
    for (int nf = 0; nf < 4; ++nf)
#pragma unroll
      for (int j = 0; j < 8; ++j)
        Ps[w][j + 8 * hl][nf * 16 + n16] = f2bf(s[nf][j]);

    // O += P @ V  (keys = K dim, two 32-wide k-steps)
#pragma unroll
    for (int s2 = 0; s2 < 2; ++s2) {
      v16bf pa = frag_rowmajor(&Ps[w][n16][s2 * 32], hl);
#pragma unroll
      for (int nf = 0; nf < 4; ++nf) {
        v16bf bv = frag_rowmajor(&Vts[buf][nf * 16 + n16][s2 * 32], hl);
        o[nf] = WMMA_BF16(pa, bv, o[nf]);
      }
    }

    buf ^= 1;
    *(uint4*)&Ks[buf][lr][lc]      = kRa;
    *(uint4*)&Ks[buf][lr][lc + 8]  = kRb;
    *(uint4*)&Vts[buf][lr][lc]     = vRa;
    *(uint4*)&Vts[buf][lr][lc + 8] = vRb;
    __syncthreads();
  }

  // O / l -> attn buffer [B*T, D] bf16 (heads concatenated)
#pragma unroll
  for (int nf = 0; nf < 4; ++nf)
#pragma unroll
    for (int j = 0; j < 8; ++j) {
      float val = o[nf][j] / li[j];
      int t = qrow0 + j + 8 * hl;
      attnb[(size_t)(b * SEQ_T + t) * D_MODEL + hh * HEAD_DIM + nf * 16 + n16] = f2bf(val);
    }
}

// ---------------------------------------------------------------- output GEMM (double-buffered)
// out[4096,1024] = attn @ Wout + b ; BT is Wout^T [1024][1024] bf16, f32 output.
__global__ __launch_bounds__(256)
void gemm_out(const u16* __restrict__ A, const u16* __restrict__ BT,
              const float* __restrict__ bias, float* __restrict__ out) {
  const int K = 1024, N = D_MODEL;
  __shared__ alignas(16) u16 As[2][64][40];
  __shared__ alignas(16) u16 Bt[2][128][40];

  int tid = threadIdx.x, lane = tid & 31, w = tid >> 5;
  int hl = lane >> 4, n16 = lane & 15;
  int wm = w & 3, wn = w >> 2;
  int rowBase = blockIdx.y * 64, colBase = blockIdx.x * 128;
  int ar = tid >> 2, ac = (tid & 3) * 8;
  int bn = tid >> 1, bk = (tid & 1) * 16;

  const u16* aPtr = A  + (size_t)(rowBase + ar) * K + ac;
  const u16* bPtr = BT + (size_t)(colBase + bn) * K + bk;

  uint4 aR  = *(const uint4*)aPtr;
  uint4 bR0 = *(const uint4*)bPtr;
  uint4 bR1 = *(const uint4*)(bPtr + 8);
  *(uint4*)&As[0][ar][ac]     = aR;
  *(uint4*)&Bt[0][bn][bk]     = bR0;
  *(uint4*)&Bt[0][bn][bk + 8] = bR1;
  __syncthreads();

  v8f acc[4] = {};
  int buf = 0;
  for (int k0 = 0; k0 < K; k0 += 32) {
    int kn = (k0 + 32 < K) ? (k0 + 32) : 0;
    aR  = *(const uint4*)(aPtr + kn);
    bR0 = *(const uint4*)(bPtr + kn);
    bR1 = *(const uint4*)(bPtr + kn + 8);

    v16bf a = frag_rowmajor(&As[buf][wm * 16 + n16][0], hl);
#pragma unroll
    for (int nf = 0; nf < 4; ++nf) {
      v16bf bfv = frag_rowmajor(&Bt[buf][wn * 64 + nf * 16 + n16][0], hl);
      acc[nf] = WMMA_BF16(a, bfv, acc[nf]);
    }

    buf ^= 1;
    *(uint4*)&As[buf][ar][ac]     = aR;
    *(uint4*)&Bt[buf][bn][bk]     = bR0;
    *(uint4*)&Bt[buf][bn][bk + 8] = bR1;
    __syncthreads();
  }

#pragma unroll
  for (int nf = 0; nf < 4; ++nf) {
    int col = colBase + wn * 64 + nf * 16 + n16;
    float bv = bias[col];
#pragma unroll
    for (int j = 0; j < 8; ++j) {
      int row = rowBase + wm * 16 + j + 8 * hl;
      out[(size_t)row * N + col] = acc[nf][j] + bv;
    }
  }
}

// ---------------------------------------------------------------- launcher
extern "C" void kernel_launch(void* const* d_in, const int* in_sizes, int n_in,
                              void* d_out, int out_size, void* d_ws, size_t ws_size,
                              hipStream_t stream) {
  (void)in_sizes; (void)n_in; (void)out_size; (void)ws_size;
  const float* x      = (const float*)d_in[0];
  const float* bscore = (const float*)d_in[1];
  const float* Wqkv   = (const float*)d_in[2];
  const float* bqkv   = (const float*)d_in[3];
  const float* Wout   = (const float*)d_in[4];
  const float* bout   = (const float*)d_in[5];
  float* out = (float*)d_out;

  const size_t nX  = (size_t)BATCH * SEQ_T * D_MODEL;
  const size_t nWq = (size_t)D_MODEL * 3 * D_MODEL;
  const size_t nWo = (size_t)D_MODEL * D_MODEL;
  const size_t nH  = (size_t)BATCH * NUM_HEADS * SEQ_T * HEAD_DIM;

  u16* Xb   = (u16*)d_ws;
  u16* WqbT = Xb + nX;       // Wqkv^T [3072][1024] bf16
  u16* WobT = WqbT + nWq;    // Wout^T [1024][1024] bf16
  u16* Qb   = WobT + nWo;    // [B,H,T,64]
  u16* Kb   = Qb + nH;       // [B,H,T,64]
  u16* Vtb  = Kb + nH;       // [B,H,64,T] (transposed)
  u16* Ab   = Vtb + nH;      // attn out [B*T, D] bf16

  cvt_f32_bf16<<<(int)((nX + 255) / 256), 256, 0, stream>>>(x, Xb, (int)nX);
  {
    dim3 g(3 * D_MODEL / 32, D_MODEL / 32);   // 96 x 32
    cvt_transpose<<<g, 256, 0, stream>>>(Wqkv, WqbT, D_MODEL, 3 * D_MODEL);
  }
  {
    dim3 g(D_MODEL / 32, D_MODEL / 32);       // 32 x 32
    cvt_transpose<<<g, 256, 0, stream>>>(Wout, WobT, D_MODEL, D_MODEL);
  }

  dim3 g1(3 * D_MODEL / 128, BATCH * SEQ_T / 64);  // 24 x 64
  gemm_qkv<<<g1, 256, 0, stream>>>(Xb, WqbT, bqkv, Qb, Kb, Vtb);

  flash_attn<<<BATCH * NUM_HEADS * (SEQ_T / 128), 256, 0, stream>>>(Qb, Kb, Vtb, bscore, Ab);

  dim3 g2(D_MODEL / 128, BATCH * SEQ_T / 64);      // 8 x 64
  gemm_out<<<g2, 256, 0, stream>>>(Ab, WobT, bout, out);
}
